// Get_Correlation_31284541784124
// MI455X (gfx1250) — matrix operand order repaired and checked by reading, compiled, not run on hardware
//
#include <hip/hip_runtime.h>
#include <hip/hip_bf16.h>

// ---------------------------------------------------------------------------
// Problem constants (B,C,T,H,W = 2,64,32,28,28; CR = 4)
// ---------------------------------------------------------------------------
#define Bn   2
#define Cn   64
#define Tn   32
#define Hn   28
#define Wn   28
#define HWn  (Hn*Wn)           // 784
#define CST  (Tn*HWn)          // 25088  (channel stride in [B,C,T,P])
#define NPOS (Bn*Tn*HWn)       // 50176  (b,t,p positions)
#define CRn  4
#define PBLK 49                // 784/16 query row-blocks per (b,t)
#define NELT ((size_t)Bn*Cn*CST)  // 3,211,264

typedef __attribute__((ext_vector_type(16))) __bf16 v16bf;
typedef __attribute__((ext_vector_type(8)))  __bf16 v8bf;
typedef __attribute__((ext_vector_type(8)))  float  v8f;

__device__ __forceinline__ float fast_sigmoid_m05(float v) {
  // sigmoid(v) - 0.5 with v_rcp_f32 instead of IEEE division
  return __builtin_amdgcn_rcpf(1.f + __expf(-v)) - 0.5f;
}

// ---------------------------------------------------------------------------
// Kernel 1: pointwise convs + bf16 relayout.  One block (64 thr) per position.
//   x2 = w_down2·x  -> x2b (bf16, [B,C,T,P]) and x2t (bf16, [B,T,P,C])
//   x              -> xt  (bf16, [B,T,P,C])
//   xr = w_down·x  -> fp32 [B,CR,T,P]
// ---------------------------------------------------------------------------
__global__ void k_pointwise(const float* __restrict__ x,
                            const float* __restrict__ w_down2,
                            const float* __restrict__ w_down,
                            __bf16* __restrict__ xt,
                            __bf16* __restrict__ x2b,
                            __bf16* __restrict__ x2t,
                            float* __restrict__ xr) {
  __shared__ float xs[Cn];
  const int c  = threadIdx.x;
  const int bb = blockIdx.x / (Tn*HWn);
  const int tp = blockIdx.x % (Tn*HWn);
  const size_t base = (size_t)bb*Cn*CST + tp;

  const float xv = x[base + (size_t)c*CST];
  xs[c] = xv;
  xt[(size_t)blockIdx.x*Cn + c] = (__bf16)xv;        // transposed copy of x
  __syncthreads();

  float acc = 0.f;
  #pragma unroll 8
  for (int k = 0; k < Cn; ++k) acc += w_down2[c*Cn + k] * xs[k];
  const __bf16 ab = (__bf16)acc;
  x2b[base + (size_t)c*CST]        = ab;             // [B,C,T,P]
  x2t[(size_t)blockIdx.x*Cn + c]   = ab;             // [B,T,P,C]

  if (c < CRn) {
    float a = 0.f;
    #pragma unroll 8
    for (int k = 0; k < Cn; ++k) a += w_down[c*Cn + k] * xs[k];
    xr[(size_t)bb*CRn*CST + (size_t)c*CST + tp] = a;
  }
}

// ---------------------------------------------------------------------------
// Kernel 2: three dilated depthwise (9,3,3) convs on CR=4 channels.
// ---------------------------------------------------------------------------
__global__ void k_dwconv(const float* __restrict__ xr,
                         const float* __restrict__ wsa1,
                         const float* __restrict__ wsa2,
                         const float* __restrict__ wsa3,
                         const float* __restrict__ wts,
                         float* __restrict__ agg) {
  const int idx = blockIdx.x * blockDim.x + threadIdx.x;
  if (idx >= NPOS) return;
  const int w = idx % Wn;
  const int h = (idx / Wn) % Hn;
  const int t = (idx / HWn) % Tn;
  const int b = idx / (Tn*HWn);
  const float wt[3] = { wts[0], wts[1], wts[2] };
  const float* wsa[3] = { wsa1, wsa2, wsa3 };

  for (int r = 0; r < CRn; ++r) {
    const float* base = xr + (size_t)(b*CRn + r)*CST;
    float s = 0.f;
    #pragma unroll
    for (int d = 1; d <= 3; ++d) {
      const float* wk = wsa[d-1] + r*81;             // [9][3][3]
      float sd = 0.f;
      for (int kt = 0; kt < 9; ++kt) {
        const int tt = t + kt - 4;
        if (tt < 0 || tt >= Tn) continue;
        #pragma unroll
        for (int kh = 0; kh < 3; ++kh) {
          const int hh = h + (kh-1)*d;
          if (hh < 0 || hh >= Hn) continue;
          #pragma unroll
          for (int kw = 0; kw < 3; ++kw) {
            const int ww = w + (kw-1)*d;
            if (ww < 0 || ww >= Wn) continue;
            sd += base[(tt*Hn + hh)*Wn + ww] * wk[(kt*3 + kh)*3 + kw];
          }
        }
      }
      s += sd * wt[d-1];
    }
    agg[(size_t)(b*CRn + r)*CST + t*HWn + h*Wn + w] = s;
  }
}

// ---------------------------------------------------------------------------
// Kernel 3: gate[b,c,pos] = sigmoid( w_back[c,:]·agg[:,pos] ) - 0.5
// ---------------------------------------------------------------------------
__global__ void k_gate(const float* __restrict__ agg,
                       const float* __restrict__ w_back,
                       float* __restrict__ gate) {
  const int idx = blockIdx.x * blockDim.x + threadIdx.x;
  if (idx >= (int)NELT) return;
  const int tp = idx % CST;
  const int bc = idx / CST;
  const int c  = bc % Cn;
  const int b  = bc / Cn;
  float s = 0.f;
  #pragma unroll
  for (int r = 0; r < CRn; ++r)
    s += w_back[c*CRn + r] * agg[(size_t)(b*CRn + r)*CST + tp];
  gate[idx] = fast_sigmoid_m05(s);
}

// ---------------------------------------------------------------------------
// Kernel 4: fused non-local block (flash-attention style, bf16 WMMA).
// One wave per (b, t, 16-row query block).  Per temporal-shift branch:
//   phase 1: aff tile = Q(16xC) x K(Cx16s) via v_wmma_f32_16x16x32_bf16,
//            g = (sigmoid(aff)-0.5)*w2[br]  ->  LDS tile [16p][128s] bf16
//   phase 2: o[c,p] += K(16c x 32s) x g^T   (4 c-tiles x 4 k-steps)
// All fragments are loaded with 16/32-byte contiguous vector loads from the
// bf16 relayout buffers (no per-element gathers, no inline cvt).
// ---------------------------------------------------------------------------
__global__ __launch_bounds__(128) void k_nonlocal(
    const __bf16* __restrict__ xt,    // [B,T,P,C]
    const __bf16* __restrict__ x2b,   // [B,C,T,P] (+64 elem slack)
    const __bf16* __restrict__ x2t,   // [B,T,P,C]
    const float*  __restrict__ gate,
    const float*  __restrict__ w2,    // weights2 (uses [0],[1])
    float* __restrict__ out) {

  __shared__ __align__(32) __bf16 gtile[4][16*128];  // per-wave gated-aff tile

  const int lane = threadIdx.x & 31;
  const int wid  = threadIdx.x >> 5;
  const int task = blockIdx.x * 4 + wid;             // 0 .. 3135 (exact)
  const int pb = task % PBLK;
  const int t  = (task / PBLK) % Tn;
  const int b  = task / (PBLK * Tn);
  const int p0 = pb * 16;

  const int half = lane >> 4;
  const int lr   = lane & 15;

  // ---- Q fragments (A layout: M=p, K=c), four 16B contiguous runs ----
  const __bf16* qp = xt + ((size_t)(b*Tn + t)*HWn + p0 + lr)*Cn;
  const v8bf q00 = *(const v8bf*)(qp +      16*half);     // c:  0..31 lo half
  const v8bf q01 = *(const v8bf*)(qp +  8 + 16*half);
  const v8bf q10 = *(const v8bf*)(qp + 32 + 16*half);     // c: 32..63
  const v8bf q11 = *(const v8bf*)(qp + 40 + 16*half);
  const v16bf a0 = __builtin_shufflevector(q00, q01, 0,1,2,3,4,5,6,7,
                                           8,9,10,11,12,13,14,15);
  const v16bf a1 = __builtin_shufflevector(q10, q11, 0,1,2,3,4,5,6,7,
                                           8,9,10,11,12,13,14,15);

  v8f o[4] = {};                       // features1 accumulators

  const int   tsh[2] = { (t + 1 < Tn) ? t + 1 : Tn - 1,  (t > 0) ? t - 1 : 0 };
  const float gsc[2] = { w2[0], w2[1] };

  for (int br = 0; br < 2; ++br) {
    const __bf16* KT = x2t + (size_t)(b*Tn + tsh[br])*HWn*Cn;          // [P][C]
    const __bf16* KBb = x2b + (size_t)b*Cn*CST + (size_t)tsh[br]*HWn;  // [C][P]
    const float gw = gsc[br];

    for (int ch = 0; ch < 7; ++ch) {                 // 7 s-chunks of 112
      const int s0 = ch * 112;

      // ---------- phase 1: affinity + sigmoid gate -> LDS ----------
      #pragma unroll
      for (int nt = 0; nt < 7; ++nt) {
        const __bf16* kp = KT + (size_t)(s0 + nt*16 + lr)*Cn;   // one s-column
        const v16bf b0 = *(const v16bf*)(kp + 16*half);         // c 0..31
        const v16bf b1 = *(const v16bf*)(kp + 32 + 16*half);    // c 32..63
        v8f z = {};
        v8f acc = __builtin_amdgcn_wmma_f32_16x16x32_bf16(
            false, a1, false, b1, (short)0, z, false, false);
        acc = __builtin_amdgcn_wmma_f32_16x16x32_bf16(
            false, a0, false, b0, (short)0, acc, false, false);
        #pragma unroll
        for (int i = 0; i < 8; ++i)                  // D: row i+8*half, col lr
          gtile[wid][(i + 8*half)*128 + nt*16 + lr] =
              (__bf16)(fast_sigmoid_m05(acc[i]) * gw);
      }
      // zero-pad columns 112..127 (half-empty 4th k-step contributes 0)
      #pragma unroll
      for (int i = 0; i < 8; ++i)
        gtile[wid][(i + 8*half)*128 + 112 + lr] = (__bf16)0.f;

      asm volatile("s_wait_dscnt 0" ::: "memory");   // LDS write -> read, same wave

      // ---------- phase 2: o[c,p] += K * g^T ----------
      #pragma unroll
      for (int ks = 0; ks < 4; ++ks) {
        // B operand: B[k=s_local][n=p] = g[p][s_local]; contiguous 16 bf16
        const v16bf bg =
            *(const v16bf*)&gtile[wid][lr*128 + ks*32 + 16*half];
        #pragma unroll
        for (int mt = 0; mt < 4; ++mt) {
          // A operand: M=c (tile mt), K=s; two 16B contiguous runs
          const __bf16* ap = KBb + (size_t)(mt*16 + lr)*CST + s0 + ks*32;
          const v8bf lo = *(const v8bf*)(ap +      16*half);
          const v8bf hi = *(const v8bf*)(ap +  8 + 16*half);  // may read slack
          const v16bf a2 = __builtin_shufflevector(lo, hi, 0,1,2,3,4,5,6,7,
                                                   8,9,10,11,12,13,14,15);
          o[mt] = __builtin_amdgcn_wmma_f32_16x16x32_bf16(
              false, a2, false, bg, (short)0, o[mt], false, false);
        }
      }
    }
  }

  // ---------- epilogue: multiply by conv-branch gate, store ----------
  const float* GB = gate + (size_t)b*Cn*CST + (size_t)t*HWn;
  float*       OB = out  + (size_t)b*Cn*CST + (size_t)t*HWn;
  #pragma unroll
  for (int mt = 0; mt < 4; ++mt)
    #pragma unroll
    for (int i = 0; i < 8; ++i) {
      const int c = mt*16 + i + 8*half;
      const size_t idx = (size_t)c * CST + (p0 + lr);
      OB[idx] = o[mt][i] * GB[idx];
    }
}

// ---------------------------------------------------------------------------
// Launch.  Workspace layout (~34 MB):
//   xr (fp32) | agg (fp32) | gate (fp32) | xt (bf16) | x2b (bf16+slack) | x2t
// ---------------------------------------------------------------------------
extern "C" void kernel_launch(void* const* d_in, const int* in_sizes, int n_in,
                              void* d_out, int out_size, void* d_ws, size_t ws_size,
                              hipStream_t stream) {
  const float* x        = (const float*)d_in[0];
  const float* w_down   = (const float*)d_in[1];
  const float* w_down2  = (const float*)d_in[2];
  const float* w_sa1    = (const float*)d_in[3];
  const float* w_sa2    = (const float*)d_in[4];
  const float* w_sa3    = (const float*)d_in[5];
  const float* weights  = (const float*)d_in[6];
  const float* weights2 = (const float*)d_in[7];
  const float* w_back   = (const float*)d_in[8];

  const size_t nSmall = (size_t)Bn*CRn*CST;          //   200,704 floats
  char* wsb = (char*)d_ws;
  float*  xr   = (float*)wsb;
  float*  agg  = (float*)(wsb + nSmall*4);
  float*  gate = (float*)(wsb + 2*nSmall*4);
  __bf16* xt   = (__bf16*)(wsb + 2*nSmall*4 + NELT*4);
  __bf16* x2b  = xt + NELT;
  __bf16* x2t  = x2b + NELT + 64;                    // 64-elem slack for tail

  k_pointwise<<<NPOS, Cn, 0, stream>>>(x, w_down2, w_down, xt, x2b, x2t, xr);
  k_dwconv<<<(NPOS + 127) / 128, 128, 0, stream>>>(xr, w_sa1, w_sa2, w_sa3,
                                                   weights, agg);
  k_gate<<<((int)NELT + 255) / 256, 256, 0, stream>>>(agg, w_back, gate);
  k_nonlocal<<<(Bn*Tn*PBLK) / 4, 128, 0, stream>>>(xt, x2b, x2t, gate,
                                                   weights2, (float*)d_out);
}
